// MKGAT_13245679141184
// MI455X (gfx1250) — compile-verified, matrix operand order checked by reading, not run on hardware
//
#include <hip/hip_runtime.h>
#include <hip/hip_bf16.h>
#include <stdint.h>

// ---------------- problem constants (match reference) ----------------
#define N_USERS_C   50000
#define N_ITEMS_C   10000
#define N_NODES_C   100000
#define EMB_C       64
#define NNZ_TOT     1600000   // == 50000 waves * 32 edges, exact
#define VIS_DIM_C   2048
#define TXT_DIM_C   300
#define OUT_STRIDE  192       // emb * (2 layers + 1)

typedef __attribute__((ext_vector_type(16))) __bf16 v16bf;
typedef __attribute__((ext_vector_type(8)))  float  v8f;

struct AV32 { uint4 a, b; };        // 32 bytes -> v16bf
struct BV32 { uint4 a, b; };        // 32 bytes -> v16bf

// A-source modes for the GEMM LDS fill
#define AM_F32    0   // fp32 [M,lda], convert on fill
#define AM_BF16   1   // bf16 [M,lda], straight copy
#define AM_CONCAT 2   // K=128: cols 0..63 from f32 Av (stride lda), 64..127 from f32 Av2 (stride 64)
#define AM_AVG    3   // 0.5*(Av + Av2), both f32 stride lda

__device__ __forceinline__ unsigned short f2bf(float f) {
  unsigned u = __float_as_uint(f);
  u += 0x7FFFu + ((u >> 16) & 1u);   // round-to-nearest-even
  return (unsigned short)(u >> 16);
}
__device__ __forceinline__ unsigned pk2(float lo, float hi) {
  return (unsigned)f2bf(lo) | ((unsigned)f2bf(hi) << 16);
}

// ---------------- tiled WMMA GEMM:  C = act(A[M,K] * W[N,Kpad]^T + b) ----------------
// One wave -> one 16x16 C tile. WPB waves/block share a 16x64 A slab staged in LDS.
// K loop unrolled by 64 -> 2 WMMAs per barrier pair. Kpad % 64 == 0 (weights zero-padded),
// so B loads are unconditional b128s. Grids cover N exactly -> no wave guards, EXEC all ones.
// ACT: 0=none, 1=relu, 2=leaky(0.01). K_TAIL: real K < Kpad (guard A fill only).
template<int ACT, int AMODE, bool OUT_BF16, int WPB, bool K_TAIL>
__global__ __launch_bounds__(WPB * 32)
void gemm_wmma(const void* __restrict__ Av, int lda,
               const void* __restrict__ Av2,            // second A source (CONCAT/AVG)
               const unsigned short* __restrict__ W,    // bf16 [N, Kpad] row-major, zero-padded
               const float* __restrict__ bias,
               void* __restrict__ Cv, int ldc,
               int K, int Kpad)
{
  __shared__ __align__(16) unsigned ldsA[16 * 36];   // 16 rows x 32 k-pairs, padded to 36

  const int tid  = threadIdx.x;
  const int lane = tid & 31;
  const int wave = tid >> 5;
  const int h    = lane >> 4;        // lane half (K sub-block select)
  const int n    = lane & 15;        // tile column (B) / tile row (A)
  const int mBase = blockIdx.x << 4;
  const int nCol  = blockIdx.y * (WPB * 16) + (wave << 4);

  const float bval = bias[nCol + n];
  v8f acc = {0.f, 0.f, 0.f, 0.f, 0.f, 0.f, 0.f, 0.f};

  const unsigned short* wrow = W + (size_t)(nCol + n) * Kpad;

  for (int kk = 0; kk < Kpad; kk += 64) {
    // ---- stage A slab (16 rows x 64 K as bf16 pairs) into LDS ----
    __syncthreads();
#pragma unroll
    for (int s = tid; s < 512; s += WPB * 32) {
      const int row = s >> 5, kp = s & 31;
      const int kIdx = kk + (kp << 1);
      unsigned pack;
      if (K_TAIL && kIdx >= K) {
        pack = 0u;                   // zero-pad real-K tail (K even in all uses)
      } else if (AMODE == AM_BF16) {
        pack = *reinterpret_cast<const unsigned*>(
            reinterpret_cast<const unsigned short*>(Av) + (size_t)(mBase + row) * lda + kIdx);
      } else if (AMODE == AM_F32) {
        const float* ap = reinterpret_cast<const float*>(Av) + (size_t)(mBase + row) * lda + kIdx;
        pack = pk2(ap[0], ap[1]);
      } else if (AMODE == AM_CONCAT) {
        // fused [cur | nb] concat: cols 0..63 strided in output, 64..127 from nb
        const float* ap = (kIdx < EMB_C)
            ? reinterpret_cast<const float*>(Av)  + (size_t)(mBase + row) * lda   + kIdx
            : reinterpret_cast<const float*>(Av2) + (size_t)(mBase + row) * EMB_C + (kIdx - EMB_C);
        pack = pk2(ap[0], ap[1]);
      } else { // AM_AVG: fused 0.5*(Vis2+Txt2)
        const size_t o = (size_t)(mBase + row) * lda + kIdx;
        const float* ap = reinterpret_cast<const float*>(Av)  + o;
        const float* bp = reinterpret_cast<const float*>(Av2) + o;
        pack = pk2(0.5f * (ap[0] + bp[0]), 0.5f * (ap[1] + bp[1]));
      }
      ldsA[row * 36 + kp] = pack;
    }
    __syncthreads();

    // ---- two 16x16x32 WMMAs over the staged 64-K slab ----
#pragma unroll
    for (int half = 0; half < 2; ++half) {
      const unsigned* rp = &ldsA[n * 36 + half * 16];
      AV32 ar;
      ar.a = *reinterpret_cast<const uint4*>(rp + (h << 2));      // K {0..7}+h*8 (pairs)
      ar.b = *reinterpret_cast<const uint4*>(rp + 8 + (h << 2));  // K {16..23}+h*8
      const unsigned short* wp = wrow + kk + half * 32;
      BV32 br;
      br.a = *reinterpret_cast<const uint4*>(wp + (h << 3));      // unconditional b128
      br.b = *reinterpret_cast<const uint4*>(wp + 16 + (h << 3));
      v16bf afrag = __builtin_bit_cast(v16bf, ar);
      v16bf bfrag = __builtin_bit_cast(v16bf, br);
      acc = __builtin_amdgcn_wmma_f32_16x16x32_bf16(
          false, afrag, false, bfrag, (short)0, acc, false, false);
    }
  }

#pragma unroll
  for (int r = 0; r < 8; ++r) {
    float v = acc[r] + bval;
    if (ACT == 1) v = v > 0.f ? v : 0.f;
    if (ACT == 2) v = v > 0.f ? v : 0.01f * v;
    const size_t o = (size_t)(mBase + r + (h << 3)) * (size_t)ldc + nCol + n;
    if (OUT_BF16) reinterpret_cast<unsigned short*>(Cv)[o] = f2bf(v);
    else          reinterpret_cast<float*>(Cv)[o] = v;
  }
}

// ---------------- elementwise / graph kernels ----------------
// fp32 -> bf16 with zero row padding to Kpad (Kpad >= K)
__global__ void convert_pad_k(const float* __restrict__ s, unsigned short* __restrict__ d,
                              int K, int Kpad, int total /* rows*Kpad */) {
  int i = blockIdx.x * 256 + threadIdx.x;
  if (i < total) {
    int r = i / Kpad, k = i - r * Kpad;
    d[i] = (k < K) ? f2bf(s[(size_t)r * K + k]) : (unsigned short)0;
  }
}

__global__ void zero_f32_k(float* __restrict__ p, int n) {
  int i = blockIdx.x * 256 + threadIdx.x;
  if (i < n) p[i] = 0.f;
}

// out[node, 0:64] = embedding[node]  (item rows overwritten later by Wd GEMM)
__global__ void copy_embed_k(const float* __restrict__ emb, float* __restrict__ out) {
  int i = blockIdx.x * 256 + threadIdx.x;
  if (i < N_NODES_C * EMB_C) {
    int node = i >> 6, d = i & 63;
    out[(size_t)node * OUT_STRIDE + d] = emb[i];
  }
}

// SpMM scatter: nb[row] += val * cur[col]. One wave handles 32 edges; edge metadata
// loaded once per lane then broadcast via shuffles; each lane covers 2 dims per edge.
__global__ __launch_bounds__(256)
void spmm_scatter_k(const int* __restrict__ rows, const int* __restrict__ cols,
                    const float* __restrict__ vals,
                    const float* __restrict__ cur, int curStride,
                    float* __restrict__ nb) {
  const int lane = threadIdx.x & 31;
  const int waveG = blockIdx.x * 8 + (threadIdx.x >> 5);
  const int eBase = waveG * 32;                 // NNZ_TOT == gridDim*8*32 exactly
  const int e = eBase + lane;
  const int   rv = rows[e];
  const int   cv = cols[e];
  const float vv = vals[e];
  const int d = lane << 1;
#pragma unroll 4
  for (int i = 0; i < 32; ++i) {
    const int   r = __shfl(rv, i, 32);
    const int   c = __shfl(cv, i, 32);
    const float v = __shfl(vv, i, 32);
    const float* cp = cur + (size_t)c * curStride + d;
    atomicAdd(&nb[(size_t)r * EMB_C + d],     v * cp[0]);
    atomicAdd(&nb[(size_t)r * EMB_C + d + 1], v * cp[1]);
  }
}

// ---------------- host orchestration ----------------
extern "C" void kernel_launch(void* const* d_in, const int* in_sizes, int n_in,
                              void* d_out_v, int out_size, void* d_ws, size_t ws_size,
                              hipStream_t stream) {
  const int*   adj_row = (const int*)d_in[0];
  const int*   adj_col = (const int*)d_in[1];
  const float* adj_val = (const float*)d_in[2];
  const float* embedding = (const float*)d_in[3];
  const float* visf = (const float*)d_in[4];
  const float* txtf = (const float*)d_in[5];
  const float* Wv1 = (const float*)d_in[6];  const float* bv1 = (const float*)d_in[7];
  const float* Wv2 = (const float*)d_in[8];  const float* bv2 = (const float*)d_in[9];
  const float* Wt1 = (const float*)d_in[10]; const float* bt1 = (const float*)d_in[11];
  const float* Wt2 = (const float*)d_in[12]; const float* bt2 = (const float*)d_in[13];
  const float* Wd  = (const float*)d_in[14]; const float* bd  = (const float*)d_in[15];
  const float* Wc0 = (const float*)d_in[16]; const float* bc0 = (const float*)d_in[17];
  const float* Wc1 = (const float*)d_in[18]; const float* bc1 = (const float*)d_in[19];
  float* out = (float*)d_out_v;

  char* ws = (char*)d_ws;
  size_t off = 0;
  auto take = [&](size_t bytes) -> char* {
    char* p = ws + off;
    off = (off + bytes + 255) & ~(size_t)255;
    return p;
  };

  const int TXT_PAD = 320;   // ceil(300/64)*64
  unsigned short* Wv1b = (unsigned short*)take((size_t)512 * VIS_DIM_C * 2);
  unsigned short* Wv2b = (unsigned short*)take((size_t)64 * 512 * 2);
  unsigned short* Wt1b = (unsigned short*)take((size_t)256 * TXT_PAD * 2);
  unsigned short* Wt2b = (unsigned short*)take((size_t)64 * 256 * 2);
  unsigned short* Wdb  = (unsigned short*)take((size_t)64 * 64 * 2);
  unsigned short* Wc0b = (unsigned short*)take((size_t)64 * 128 * 2);
  unsigned short* Wc1b = (unsigned short*)take((size_t)64 * 128 * 2);
  unsigned short* H1b  = (unsigned short*)take((size_t)N_ITEMS_C * 512 * 2);
  unsigned short* T1b  = (unsigned short*)take((size_t)N_ITEMS_C * 256 * 2);
  float*          Vis2 = (float*)take((size_t)N_ITEMS_C * EMB_C * 4);
  float*          Txt2 = (float*)take((size_t)N_ITEMS_C * EMB_C * 4);
  float*          nb   = (float*)take((size_t)N_NODES_C * EMB_C * 4);

  auto cvt = [&](const float* s, unsigned short* d, int rows, int K, int Kpad) {
    int total = rows * Kpad;
    convert_pad_k<<<(total + 255) / 256, 256, 0, stream>>>(s, d, K, Kpad, total);
  };
  cvt(Wv1, Wv1b, 512, VIS_DIM_C, VIS_DIM_C);
  cvt(Wv2, Wv2b, 64, 512, 512);
  cvt(Wt1, Wt1b, 256, TXT_DIM_C, TXT_PAD);
  cvt(Wt2, Wt2b, 64, 256, 256);
  cvt(Wd,  Wdb,  64, 64, 64);
  cvt(Wc0, Wc0b, 64, 128, 128);
  cvt(Wc1, Wc1b, 64, 128, 128);

  const int mtI = N_ITEMS_C / 16;   // 625
  const int mtN = N_NODES_C / 16;   // 6250

  // Visual branch: H1 = relu(Vis @ Wv1^T + bv1)  [10k,512] bf16
  gemm_wmma<1, AM_F32, true, 8, false><<<dim3(mtI, 4), 256, 0, stream>>>(
      visf, VIS_DIM_C, nullptr, Wv1b, bv1, H1b, 512, VIS_DIM_C, VIS_DIM_C);
  // Vis2 = H1 @ Wv2^T + bv2  [10k,64] f32
  gemm_wmma<0, AM_BF16, false, 4, false><<<dim3(mtI, 1), 128, 0, stream>>>(
      H1b, 512, nullptr, Wv2b, bv2, Vis2, 64, 512, 512);
  // Text branch: T1 = relu(Txt @ Wt1^T + bt1)  [10k,256] bf16 (K tail 300->320)
  gemm_wmma<1, AM_F32, true, 8, true><<<dim3(mtI, 2), 256, 0, stream>>>(
      txtf, TXT_DIM_C, nullptr, Wt1b, bt1, T1b, 256, TXT_DIM_C, TXT_PAD);
  // Txt2 = T1 @ Wt2^T + bt2  [10k,64] f32
  gemm_wmma<0, AM_BF16, false, 4, false><<<dim3(mtI, 1), 128, 0, stream>>>(
      T1b, 256, nullptr, Wt2b, bt2, Txt2, 64, 256, 256);

  // ego: copy all embeddings into out[:,0:64], then overwrite item rows with
  // fused_item = (0.5*(Vis2+Txt2)) @ Wd^T + bd  (avg fused into the GEMM A fill)
  copy_embed_k<<<(N_NODES_C * EMB_C + 255) / 256, 256, 0, stream>>>(embedding, out);
  gemm_wmma<0, AM_AVG, false, 4, false><<<dim3(mtI, 1), 128, 0, stream>>>(
      Vis2, EMB_C, Txt2, Wdb, bd, out + (size_t)N_USERS_C * OUT_STRIDE, OUT_STRIDE, 64, 64);

  // Two graph-conv layers; activations live in strided slices of d_out and the
  // [cur | nb] concat is fused into the GEMM's LDS fill (no AB materialization).
  const unsigned short* Wcb[2] = { Wc0b, Wc1b };
  const float* bcb[2] = { bc0, bc1 };
  for (int l = 0; l < 2; ++l) {
    float* cur = out + (size_t)l * EMB_C;          // cols [64*l, 64*l+64)
    float* nxt = out + (size_t)(l + 1) * EMB_C;    // cols [64*(l+1), ...)
    zero_f32_k<<<(N_NODES_C * EMB_C + 255) / 256, 256, 0, stream>>>(nb, N_NODES_C * EMB_C);
    spmm_scatter_k<<<NNZ_TOT / 32 / 8, 256, 0, stream>>>(adj_row, adj_col, adj_val, cur, OUT_STRIDE, nb);
    gemm_wmma<2, AM_CONCAT, false, 4, false><<<dim3(mtN, 1), 128, 0, stream>>>(
        cur, OUT_STRIDE, nb, Wcb[l], bcb[l], nxt, OUT_STRIDE, 128, 128);
  }
}
